// EncoderLayer_3058016715067
// MI455X (gfx1250) — matrix operand order repaired and checked by reading, not compiled, not run on hardware
//
#include <hip/hip_runtime.h>
#include <cstddef>
#include <cstdint>

// ---------------------------------------------------------------------------
// Problem constants (match reference)
// ---------------------------------------------------------------------------
static constexpr int S_   = 1024;
static constexpr int B_   = 4;
static constexpr int D_   = 768;
static constexpr int H_   = 12;
static constexpr int HD_  = 64;
static constexpr int FFN_ = 3072;
static constexpr int M_   = S_ * B_;      // 4096 token rows
#define NEGV (-1e9f)

// ---------------------------------------------------------------------------
// CDNA5 WMMA types / helpers
// ---------------------------------------------------------------------------
typedef __attribute__((ext_vector_type(16))) __bf16 v16bf;
typedef __attribute__((ext_vector_type(8)))  float  v8f;
typedef __attribute__((ext_vector_type(4)))  int    v4i;

#define AS_GLOBAL __attribute__((address_space(1)))
#define AS_LOCAL  __attribute__((address_space(3)))

// gfx1250 async global->LDS copy (ASYNCcnt-tracked), with sync fallback.
#if __has_builtin(__builtin_amdgcn_global_load_async_to_lds_b128) && \
    __has_builtin(__builtin_amdgcn_s_wait_asynccnt)
#define USE_ASYNC_COPY 1
#else
#define USE_ASYNC_COPY 0
#endif

// Copy 16 bytes (8 bf16) global -> LDS. Both addresses 16B-aligned.
__device__ __forceinline__ void copy16_g2l(const __bf16* g, __bf16* l) {
#if USE_ASYNC_COPY
  __builtin_amdgcn_global_load_async_to_lds_b128(
      (AS_GLOBAL v4i*)(uintptr_t)g,
      (AS_LOCAL v4i*)(uint32_t)(uintptr_t)l, 0, 0);
#else
#pragma unroll
  for (int j = 0; j < 8; ++j) l[j] = g[j];
#endif
}

__device__ __forceinline__ void async_fence() {
#if USE_ASYNC_COPY
  __builtin_amdgcn_s_wait_asynccnt(0);
#endif
}

__device__ __forceinline__ v8f wmma_bf16(v16bf a, v16bf b, v8f c) {
  // D = A(16x32 bf16) x B(32x16 bf16) + C(16x16 f32)
  return __builtin_amdgcn_wmma_f32_16x16x32_bf16(
      /*neg_a=*/false, a, /*neg_b=*/false, b,
      /*c_mod=*/(short)0, c, /*reuse_a=*/false, /*reuse_b=*/false);
}

__device__ __forceinline__ __bf16 f2bf(float f) {
  union { float f; unsigned u; } v; v.f = f;
  unsigned r = v.u + 0x7fffu + ((v.u >> 16) & 1u);   // round-to-nearest-even
  unsigned short h = (unsigned short)(r >> 16);
  __bf16 o; __builtin_memcpy(&o, &h, 2);
  return o;
}

__device__ __forceinline__ v8f zero8() {
  v8f z;
#pragma unroll
  for (int i = 0; i < 8; ++i) z[i] = 0.f;
  return z;
}

__device__ __forceinline__ v16bf load16(const __bf16* p) {
  v16bf r;
#pragma unroll
  for (int i = 0; i < 16; ++i) r[i] = p[i];
  return r;
}

// A-matrix fragment (16x32 bf16), contiguous k order (same as the B loads):
// lane L holds row L%16; element s holds K = (L/16)*16 + s.
__device__ __forceinline__ v16bf load_a_frag(const __bf16* tile, int pitch, int lane) {
  const int row = lane & 15;
  const int kb  = (lane >> 4) << 4;
  const __bf16* p = tile + row * pitch;
  v16bf a;
#pragma unroll
  for (int j = 0; j < 8; ++j) { a[j] = p[kb + j]; a[8 + j] = p[kb + 8 + j]; }
  return a;
}

// ---------------------------------------------------------------------------
// f32 -> bf16 conversion (grid-stride)
// ---------------------------------------------------------------------------
__global__ __launch_bounds__(256) void cvt_bf16_kernel(const float* __restrict__ in,
                                                       __bf16* __restrict__ out, int n) {
  for (int i = blockIdx.x * 256 + threadIdx.x; i < n; i += gridDim.x * 256)
    out[i] = f2bf(in[i]);
}

// ---------------------------------------------------------------------------
// Generic bf16 WMMA GEMM: C(MxN,f32) = A(MxK) * B(KxN); epilogue by MODE.
//   MODE 0: out = (acc + bias) * scale          -> bf16  (QKV projections)
//   MODE 1: out = acc + bias + resid            -> f32   (O-proj, FFN2 pre-LN)
//   MODE 2: out = gelu(acc + bias)              -> bf16  (FFN1)
// Block: 256 threads = 8 waves; block tile 64(M) x 256(N); wave tile 32x64.
// Per K-step: 8 WMMAs/wave, B-fragments reused across both A sub-tiles.
// ---------------------------------------------------------------------------
template <int MODE>
__global__ __launch_bounds__(256) void gemm_bf16_kernel(
    const __bf16* __restrict__ A, const __bf16* __restrict__ Bw,
    const float* __restrict__ bias, const float* __restrict__ resid,
    float scale, __bf16* __restrict__ outb, float* __restrict__ outf,
    int M, int N, int K) {
  constexpr int PA = 40;   // LDS pitch (bf16) for A rows; 40*2=80B keeps 16B align
  constexpr int PB = 40;   // LDS pitch for transposed-B rows (one output col)
  __shared__ __bf16 ldsA[64 * PA];
  __shared__ __bf16 ldsB[256 * PB];

  const int tid  = threadIdx.x;
  const int lane = tid & 31;
  const int wave = tid >> 5;
  const int wm   = wave >> 2;          // 0..1  (M sub-block of 32 rows)
  const int wn   = wave & 3;           // 0..3  (N sub-tile of 64 cols)
  const int mBlock = blockIdx.y * 64;
  const int nBlock = blockIdx.x * 256;

  v8f acc[2][4];
#pragma unroll
  for (int s = 0; s < 2; ++s)
#pragma unroll
    for (int nj = 0; nj < 4; ++nj) acc[s][nj] = zero8();

  for (int kb = 0; kb < K; kb += 32) {
    // ---- stage A tile: 64 rows x 32 K; 16B per thread (async if available) --
    {
      const int row = tid >> 2;          // 0..63
      const int col = (tid & 3) * 8;     // 0,8,16,24
      copy16_g2l(A + (size_t)(mBlock + row) * K + kb + col,
                 &ldsA[row * PA + col]);
    }
    // ---- stage B tile transposed: ldsB[col][k], coalesced global reads ----
    {
      const __bf16* src = Bw + (size_t)kb * N + nBlock + tid;
#pragma unroll 8
      for (int kk = 0; kk < 32; ++kk) ldsB[tid * PB + kk] = src[(size_t)kk * N];
    }
    if (kb + 32 < K)  // hint next B slice into cache (global_prefetch_b8)
      __builtin_prefetch(Bw + (size_t)(kb + 32) * N + nBlock + tid, 0, 1);
    async_fence();
    __syncthreads();

    v16bf af[2];
#pragma unroll
    for (int s = 0; s < 2; ++s)
      af[s] = load_a_frag(ldsA + (wm * 32 + s * 16) * PA, PA, lane);
#pragma unroll
    for (int nj = 0; nj < 4; ++nj) {
      const __bf16* bp = ldsB + (size_t)(wn * 64 + nj * 16 + (lane & 15)) * PB +
                         ((lane >> 4) << 4);
      const v16bf bfr = load16(bp);
#pragma unroll
      for (int s = 0; s < 2; ++s) acc[s][nj] = wmma_bf16(af[s], bfr, acc[s][nj]);
    }
    __syncthreads();
  }

  // ---- epilogue: D-matrix layout (lane half -> rows v+8*(lane/16)) ----
#pragma unroll
  for (int s = 0; s < 2; ++s)
#pragma unroll
    for (int nj = 0; nj < 4; ++nj) {
      const int col = nBlock + wn * 64 + nj * 16 + (lane & 15);
#pragma unroll
      for (int v = 0; v < 8; ++v) {
        const int row = mBlock + wm * 32 + s * 16 + v + ((lane >> 4) << 3);
        float val = acc[s][nj][v];
        if (MODE == 0) {
          val = (val + bias[col]) * scale;
        } else if (MODE == 1) {
          val = val + bias[col] + resid[(size_t)row * N + col];
        } else {  // MODE == 2: exact GELU
          val += bias[col];
          val = 0.5f * val * (1.0f + erff(val * 0.70710678118654752f));
        }
        if (outf) outf[(size_t)row * N + col] = val;
        if (outb) outb[(size_t)row * N + col] = f2bf(val);
      }
    }
}

// ---------------------------------------------------------------------------
// Flash attention, bf16 WMMA, online softmax.
// Grid: (S/(16*8), B*H). Block = 8 waves; wave w owns q-tile (blockIdx.x*8+w).
// Per key-step (32 keys): 4 score WMMAs + 4 P*V WMMAs per wave.
// ---------------------------------------------------------------------------
__global__ __launch_bounds__(256) void attn_kernel(
    const __bf16* __restrict__ q, const __bf16* __restrict__ k,
    const __bf16* __restrict__ v, const unsigned char* __restrict__ pad,
    const unsigned char* __restrict__ mm, __bf16* __restrict__ attn) {
  constexpr int PK = 72;  // ldsK pitch: [key 0..31][hd 0..63]; 144B rows (16B-mult)
  constexpr int PV = 40;  // ldsV pitch: [hd 0..63][key 0..31] (transposed)
  constexpr int PP = 40;  // ldsP pitch: per-wave 16x32 P tile
  __shared__ __bf16 ldsK[32 * PK];
  __shared__ __bf16 ldsV[64 * PV];
  __shared__ __bf16 ldsP[8 * 16 * PP];

  const int tid  = threadIdx.x;
  const int lane = tid & 31;
  const int wave = tid >> 5;
  const int bh   = blockIdx.y;
  const int bidx = bh / H_;
  const int h    = bh % H_;
  const int qTile = blockIdx.x * 8 + wave;   // 0..63
  __bf16* pw = ldsP + wave * 16 * PP;

  // Q fragments (16 rows x 64 hd -> two 16x32 A-frags), loaded once.
  v16bf qf[2];
  {
    const int qs = qTile * 16 + (lane & 15);
    const __bf16* qb = q + ((size_t)qs * B_ + bidx) * D_ + h * HD_;
    const int kb8 = (lane >> 4) << 4;
#pragma unroll
    for (int c = 0; c < 2; ++c)
#pragma unroll
      for (int j = 0; j < 8; ++j) {
        qf[c][j]     = qb[c * 32 + kb8 + j];
        qf[c][8 + j] = qb[c * 32 + kb8 + 8 + j];
      }
  }

  float mrun[8], lrun[8];
  v8f o[4];
#pragma unroll
  for (int i = 0; i < 8; ++i) { mrun[i] = -__builtin_inff(); lrun[i] = 0.f; }
#pragma unroll
  for (int nj = 0; nj < 4; ++nj) o[nj] = zero8();

  for (int keyBase = 0; keyBase < S_; keyBase += 32) {
    __syncthreads();
    // ---- cooperative stage: K tile async (contiguous), V tile transposed ----
    {
      const int key = tid >> 3;            // 0..31
      const int hd0 = (tid & 7) * 8;       // 0..56
      const size_t rbase = ((size_t)(keyBase + key) * B_ + bidx) * D_ + h * HD_;
      copy16_g2l(k + rbase + hd0, &ldsK[key * PK + hd0]);
      const __bf16* vs = v + rbase + hd0;
#pragma unroll
      for (int j = 0; j < 8; ++j) ldsV[(hd0 + j) * PV + key] = vs[j];
    }
    async_fence();
    __syncthreads();

    // ---- scores: S(16x32) = Q(16x64) * K^T(64x32), two 16x16 tiles ----
    v8f s1 = zero8(), s2 = zero8();
    {
      const int kofs = (lane >> 4) << 4;
#pragma unroll
      for (int c = 0; c < 2; ++c) {
        s1 = wmma_bf16(qf[c], load16(&ldsK[(lane & 15) * PK + c * 32 + kofs]), s1);
        s2 = wmma_bf16(qf[c], load16(&ldsK[(16 + (lane & 15)) * PK + c * 32 + kofs]), s2);
      }
    }

    // ---- masks + online softmax (rows live across 16-lane halves) ----
    const int keyc = keyBase + (lane & 15);
    const bool pm0 = pad[bidx * S_ + keyc] != 0;
    const bool pm1 = pad[bidx * S_ + keyc + 16] != 0;
#pragma unroll
    for (int vv = 0; vv < 8; ++vv) {
      const int qs = qTile * 16 + vv + ((lane >> 4) << 3);
      const unsigned char* mrow = mm + ((size_t)bidx * S_ + qs) * S_;
      float a = s1[vv]; if (pm0 || mrow[keyc])      a = NEGV;
      float b = s2[vv]; if (pm1 || mrow[keyc + 16]) b = NEGV;

      float mx = fmaxf(a, b);
      mx = fmaxf(mx, __shfl_xor(mx, 1, 16));
      mx = fmaxf(mx, __shfl_xor(mx, 2, 16));
      mx = fmaxf(mx, __shfl_xor(mx, 4, 16));
      mx = fmaxf(mx, __shfl_xor(mx, 8, 16));

      const float mn   = fmaxf(mrun[vv], mx);
      const float corr = __expf(mrun[vv] - mn);
      const float p1 = __expf(a - mn);
      const float p2 = __expf(b - mn);
      float ts = p1 + p2;
      ts += __shfl_xor(ts, 1, 16);
      ts += __shfl_xor(ts, 2, 16);
      ts += __shfl_xor(ts, 4, 16);
      ts += __shfl_xor(ts, 8, 16);

      lrun[vv] = lrun[vv] * corr + ts;
      mrun[vv] = mn;
#pragma unroll
      for (int nj = 0; nj < 4; ++nj) o[nj][vv] *= corr;

      const int prow = vv + ((lane >> 4) << 3);
      pw[prow * PP + (lane & 15)]      = f2bf(p1);
      pw[prow * PP + 16 + (lane & 15)] = f2bf(p2);
    }

    // wave-local P tile: order LDS writes before A-frag reload
    asm volatile("s_wait_dscnt 0x0" ::: "memory");

    // ---- O += P(16x32) * V(32x64) ----
    const v16bf pf = load_a_frag(pw, PP, lane);
#pragma unroll
    for (int nj = 0; nj < 4; ++nj) {
      const __bf16* vp = ldsV + (size_t)(nj * 16 + (lane & 15)) * PV +
                         ((lane >> 4) << 4);
      o[nj] = wmma_bf16(pf, load16(vp), o[nj]);
    }
  }

  // ---- normalize and write attn output (bf16, token-major (s,b,h,hd)) ----
#pragma unroll
  for (int nj = 0; nj < 4; ++nj)
#pragma unroll
    for (int vv = 0; vv < 8; ++vv) {
      const int qs = qTile * 16 + vv + ((lane >> 4) << 3);
      attn[((size_t)qs * B_ + bidx) * D_ + h * HD_ + nj * 16 + (lane & 15)] =
          f2bf(o[nj][vv] / lrun[vv]);
    }
}

// ---------------------------------------------------------------------------
// Row LayerNorm over D=768; one block per row. Optional bf16 mirror output.
// ---------------------------------------------------------------------------
__global__ __launch_bounds__(256) void ln_kernel(
    const float* __restrict__ y, const float* __restrict__ g,
    const float* __restrict__ bta, float* __restrict__ outf,
    __bf16* __restrict__ outb) {
  __shared__ float red[256];
  const int row = blockIdx.x, tid = threadIdx.x;
  const float* yr = y + (size_t)row * D_;

  float s = 0.f;
  for (int d = tid; d < D_; d += 256) s += yr[d];
  red[tid] = s; __syncthreads();
  for (int off = 128; off; off >>= 1) {
    if (tid < off) red[tid] += red[tid + off];
    __syncthreads();
  }
  const float mu = red[0] * (1.0f / D_);
  __syncthreads();

  float s2 = 0.f;
  for (int d = tid; d < D_; d += 256) { float t = yr[d] - mu; s2 += t * t; }
  red[tid] = s2; __syncthreads();
  for (int off = 128; off; off >>= 1) {
    if (tid < off) red[tid] += red[tid + off];
    __syncthreads();
  }
  const float rs = rsqrtf(red[0] * (1.0f / D_) + 1e-5f);

  for (int d = tid; d < D_; d += 256) {
    const float val = (yr[d] - mu) * rs * g[d] + bta[d];
    outf[(size_t)row * D_ + d] = val;
    if (outb) outb[(size_t)row * D_ + d] = f2bf(val);
  }
}

// ---------------------------------------------------------------------------
// Host-side orchestration
// ---------------------------------------------------------------------------
extern "C" void kernel_launch(void* const* d_in, const int* in_sizes, int n_in,
                              void* d_out, int out_size, void* d_ws, size_t ws_size,
                              hipStream_t stream) {
  (void)in_sizes; (void)n_in; (void)out_size; (void)ws_size;

  const float* x   = (const float*)d_in[0];
  const unsigned char* pad = (const unsigned char*)d_in[1];
  const unsigned char* mmm = (const unsigned char*)d_in[2];
  const float* Wq = (const float*)d_in[3];  const float* bq = (const float*)d_in[4];
  const float* Wk = (const float*)d_in[5];  const float* bk = (const float*)d_in[6];
  const float* Wv = (const float*)d_in[7];  const float* bv = (const float*)d_in[8];
  const float* Wo = (const float*)d_in[9];  const float* bo = (const float*)d_in[10];
  const float* ln1g = (const float*)d_in[11]; const float* ln1b = (const float*)d_in[12];
  const float* W1 = (const float*)d_in[13]; const float* b1 = (const float*)d_in[14];
  const float* W2 = (const float*)d_in[15]; const float* b2 = (const float*)d_in[16];
  const float* ln2g = (const float*)d_in[17]; const float* ln2b = (const float*)d_in[18];
  float* out = (float*)d_out;

  const size_t MD = (size_t)M_ * D_;      // 3,145,728
  const size_t DD = (size_t)D_ * D_;      //   589,824
  const size_t DF = (size_t)D_ * FFN_;    // 2,359,296
  const size_t MF = (size_t)M_ * FFN_;    // 12,582,912

  char* w = (char*)d_ws;
  auto alloc = [&](size_t bytes) -> void* {
    void* p = (void*)w;
    w += (bytes + 255) & ~(size_t)255;
    return p;
  };
  __bf16* x_bf  = (__bf16*)alloc(MD * 2);  // reused as attn output after QKV
  __bf16* q_bf  = (__bf16*)alloc(MD * 2);  // reused as x1_bf after attention
  __bf16* k_bf  = (__bf16*)alloc(MD * 2);
  __bf16* v_bf  = (__bf16*)alloc(MD * 2);
  __bf16* wq_bf = (__bf16*)alloc(DD * 2);
  __bf16* wk_bf = (__bf16*)alloc(DD * 2);
  __bf16* wv_bf = (__bf16*)alloc(DD * 2);
  __bf16* wo_bf = (__bf16*)alloc(DD * 2);
  __bf16* w1_bf = (__bf16*)alloc(DF * 2);
  __bf16* w2_bf = (__bf16*)alloc(DF * 2);
  __bf16* h_bf  = (__bf16*)alloc(MF * 2);
  float*  y1    = (float*)alloc(MD * 4);   // reused as y2 pre-LN2
  float*  x1f   = (float*)alloc(MD * 4);
  __bf16* attn_bf = x_bf;
  __bf16* x1_bf   = q_bf;

  // 1) fp32 -> bf16 conversions (activations + all weights)
  auto cvt = [&](const float* src, __bf16* dst, size_t n) {
    int blocks = (int)((n + 1023) / 1024);
    if (blocks > 2048) blocks = 2048;
    cvt_bf16_kernel<<<blocks, 256, 0, stream>>>(src, dst, (int)n);
  };
  cvt(x,  x_bf,  MD);
  cvt(Wq, wq_bf, DD); cvt(Wk, wk_bf, DD);
  cvt(Wv, wv_bf, DD); cvt(Wo, wo_bf, DD);
  cvt(W1, w1_bf, DF); cvt(W2, w2_bf, DF);

  const float qscale = 0.125f;  // HD^-0.5 = 1/8
  const dim3 blk(256);
  const dim3 gD(D_ / 256, M_ / 64);     // (3,64)
  const dim3 gF(FFN_ / 256, M_ / 64);   // (12,64)

  // 2) QKV projections: (x@W + b) [* scale for Q] -> bf16
  gemm_bf16_kernel<0><<<gD, blk, 0, stream>>>(x_bf, wq_bf, bq, nullptr, qscale,
                                              q_bf, nullptr, M_, D_, D_);
  gemm_bf16_kernel<0><<<gD, blk, 0, stream>>>(x_bf, wk_bf, bk, nullptr, 1.0f,
                                              k_bf, nullptr, M_, D_, D_);
  gemm_bf16_kernel<0><<<gD, blk, 0, stream>>>(x_bf, wv_bf, bv, nullptr, 1.0f,
                                              v_bf, nullptr, M_, D_, D_);

  // 3) flash attention (writes attn_bf; x_bf no longer needed)
  attn_kernel<<<dim3(S_ / 128, B_ * H_), blk, 0, stream>>>(q_bf, k_bf, v_bf,
                                                           pad, mmm, attn_bf);

  // 4) output projection + residual(x) -> y1 (f32), then LN1 -> x1 (f32+bf16)
  gemm_bf16_kernel<1><<<gD, blk, 0, stream>>>(attn_bf, wo_bf, bo, x, 1.0f,
                                              nullptr, y1, M_, D_, D_);
  ln_kernel<<<M_, blk, 0, stream>>>(y1, ln1g, ln1b, x1f, x1_bf);

  // 5) FFN: gelu(x1@W1+b1) -> h (bf16); h@W2+b2 + x1 -> y2 (f32); LN2 -> out
  gemm_bf16_kernel<2><<<gF, blk, 0, stream>>>(x1_bf, w1_bf, b1, nullptr, 1.0f,
                                              h_bf, nullptr, M_, FFN_, D_);
  gemm_bf16_kernel<1><<<gD, blk, 0, stream>>>(h_bf, w2_bf, b2, x1f, 1.0f,
                                              nullptr, y1, M_, D_, FFN_);
  ln_kernel<<<M_, blk, 0, stream>>>(y1, ln2g, ln2b, out, nullptr);
}
